// CombinedSymptomMetricGNN_86268713108199
// MI455X (gfx1250) — matrix operand order repaired and checked by reading, not compiled, not run on hardware
//
#include <hip/hip_runtime.h>
#include <hip/hip_bf16.h>
#include <math.h>

#define NNODES 50000
#define INC    128
#define HIDC   64
#define NHEADS 4
#define H1OUT  (NHEADS * HIDC)   // 256
#define OUTC   4
#define NGRAPH 64
#define NEG_SLOPE 0.2f

typedef __attribute__((ext_vector_type(16))) __bf16 v16bf;
typedef __attribute__((ext_vector_type(8)))  float  v8f;

// ---------------- utility kernels ----------------

__global__ void fill_f32(float* p, float v, int n) {
    int i = blockIdx.x * blockDim.x + threadIdx.x;
    if (i < n) p[i] = v;
}

__global__ void fill_i32(int* p, int v, int n) {
    int i = blockIdx.x * blockDim.x + threadIdx.x;
    if (i < n) p[i] = v;
}

__global__ void cast_f32_bf16(const float* __restrict__ src, __bf16* __restrict__ dst, int n) {
    int i = blockIdx.x * blockDim.x + threadIdx.x;
    if (i < n) dst[i] = (__bf16)src[i];
}

// dst[c*R + r] = (bf16) src[r*C + c]   (transpose + cast: W [R,C] -> Wt [C,R])
__global__ void transpose_cast(const float* __restrict__ src, __bf16* __restrict__ dst, int R, int C) {
    int i = blockIdx.x * blockDim.x + threadIdx.x;
    if (i >= R * C) return;
    int r = i / C, c = i % C;
    dst[(size_t)c * R + r] = (__bf16)src[i];
}

// ---------------- WMMA GEMM: C[M,Ncol] = A[M,K](bf16) * Bt[Ncol,K](bf16)^T ----------------
// One wave32 per 16x16 output tile; K stepped by 32 via v_wmma_f32_16x16x32_bf16.
__global__ void gemm_bf16_wmma(const __bf16* __restrict__ A,
                               const __bf16* __restrict__ Bt,
                               float* __restrict__ C,
                               int M, int K, int Ncol) {
    const int lane  = threadIdx.x & 31;
    const int wave  = threadIdx.x >> 5;
    const int tilesN = Ncol >> 4;
    const int tilesM = M >> 4;
    const int tile = blockIdx.x * (blockDim.x >> 5) + wave;
    if (tile >= tilesM * tilesN) return;            // wave-uniform: EXEC stays all-1s

    const int tm = tile / tilesN;
    const int tn = tile % tilesN;
    const int l16  = lane & 15;
    const int half = lane >> 4;                     // K-half selector for A/B operand layout

    const __bf16* arow = A  + (size_t)(tm * 16 + l16) * K;
    const __bf16* bcol = Bt + (size_t)(tn * 16 + l16) * K;

    v8f acc = {0.f, 0.f, 0.f, 0.f, 0.f, 0.f, 0.f, 0.f};
    for (int kb = 0; kb < K; kb += 32) {
        v16bf a, b;
        const __bf16* ap = arow + kb + half * 8;    // elems 0..7: K=kb+kh*8+j ; 8..15: +16
        const __bf16* bp = bcol + kb + half * 8;    // elems 0..7: K=kb+kh*8+j ; 8..15: +16
#pragma unroll
        for (int j = 0; j < 8; ++j) { a[j] = ap[j]; a[8 + j] = ap[16 + j]; }
#pragma unroll
        for (int j = 0; j < 8; ++j) { b[j] = bp[j]; b[8 + j] = bp[16 + j]; }
        acc = __builtin_amdgcn_wmma_f32_16x16x32_bf16(false, a, false, b,
                                                      (short)0, acc, false, false);
    }
    // D layout: VGPR v -> row v + 8*half, col = lane%16
#pragma unroll
    for (int v = 0; v < 8; ++v) {
        C[(size_t)(tm * 16 + v + 8 * half) * Ncol + tn * 16 + l16] = acc[v];
    }
}

// ---------------- attention coefficients: a_src/a_dst [N, H] ----------------
__global__ void att_coeff(const float* __restrict__ h,
                          const float* __restrict__ att_src, const float* __restrict__ att_dst,
                          float* __restrict__ as_, float* __restrict__ ad_,
                          int nN, int H, int ch) {
    int i = blockIdx.x * blockDim.x + threadIdx.x;
    if (i >= nN * H) return;
    int n = i / H, hd = i % H;
    const float* hp = h + (size_t)n * H * ch + (size_t)hd * ch;
    const float* ws = att_src + hd * ch;
    const float* wd = att_dst + hd * ch;
    float s = 0.f, d = 0.f;
    for (int c = 0; c < ch; ++c) { float v = hp[c]; s += v * ws[c]; d += v * wd[c]; }
    as_[i] = s; ad_[i] = d;
}

// ---------------- edge helpers ----------------
__device__ __forceinline__ void edge_sd(const int* __restrict__ ei, int E, int e, int& s, int& d) {
    if (e < E) { s = ei[e]; d = ei[E + e]; }        // edge_index[0,:]=src, [1,:]=dst
    else       { s = d = e - E; }                   // implicit self-loops
}
__device__ __forceinline__ int   enc_f32(float f) { int i = __float_as_int(f); return i >= 0 ? i : (i ^ 0x7FFFFFFF); }
__device__ __forceinline__ float dec_f32(int e)   { int b = e >= 0 ? e : (e ^ 0x7FFFFFFF); return __int_as_float(b); }
#define ENC_NEG_INF ((int)0x807FFFFF)               // enc(-inf)

__global__ void edge_max(const int* __restrict__ ei, const float* __restrict__ as_,
                         const float* __restrict__ ad_, int* __restrict__ m,
                         int E, int nN, int H) {
    int i = blockIdx.x * blockDim.x + threadIdx.x;
    int tot = (E + nN) * H;
    if (i >= tot) return;
    int e = i / H, hd = i % H, s, d;
    edge_sd(ei, E, e, s, d);
    float v = as_[s * H + hd] + ad_[d * H + hd];
    v = v > 0.f ? v : NEG_SLOPE * v;
    atomicMax(&m[d * H + hd], enc_f32(v));
}

__global__ void edge_sum(const int* __restrict__ ei, const float* __restrict__ as_,
                         const float* __restrict__ ad_, const int* __restrict__ m,
                         float* __restrict__ denom, int E, int nN, int H) {
    int i = blockIdx.x * blockDim.x + threadIdx.x;
    int tot = (E + nN) * H;
    if (i >= tot) return;
    int e = i / H, hd = i % H, s, d;
    edge_sd(ei, E, e, s, d);
    float v = as_[s * H + hd] + ad_[d * H + hd];
    v = v > 0.f ? v : NEG_SLOPE * v;
    float ex = __expf(v - dec_f32(m[d * H + hd]));
    atomicAdd(&denom[d * H + hd], ex);
}

__global__ void edge_agg(const int* __restrict__ ei, const float* __restrict__ as_,
                         const float* __restrict__ ad_, const int* __restrict__ m,
                         const float* __restrict__ denom, const float* __restrict__ h,
                         float* __restrict__ agg, int E, int nN, int H, int ch) {
    int i = blockIdx.x * blockDim.x + threadIdx.x;
    int tot = (E + nN) * H;
    if (i >= tot) return;
    int e = i / H, hd = i % H, s, d;
    edge_sd(ei, E, e, s, d);
    float v = as_[s * H + hd] + ad_[d * H + hd];
    v = v > 0.f ? v : NEG_SLOPE * v;
    float ex = __expf(v - dec_f32(m[d * H + hd]));
    float alpha = ex / (denom[d * H + hd] + 1e-16f);
    const float* hs = h   + (size_t)s * H * ch + (size_t)hd * ch;
    float*       op = agg + (size_t)d * H * ch + (size_t)hd * ch;
    for (int c = 0; c < ch; ++c) atomicAdd(&op[c], hs[c] * alpha);
}

// bias + ELU, optionally emit bf16 copy (for next GEMM) and/or f32 copy
__global__ void bias_elu(const float* __restrict__ agg, const float* __restrict__ bias,
                         float* __restrict__ out_f32, __bf16* __restrict__ out_bf16,
                         int nN, int F) {
    int i = blockIdx.x * blockDim.x + threadIdx.x;
    if (i >= nN * F) return;
    float v = agg[i] + bias[i % F];
    v = v > 0.f ? v : expm1f(v);
    if (out_f32)  out_f32[i]  = v;
    if (out_bf16) out_bf16[i] = (__bf16)v;
}

// ---------------- pooling + heads ----------------
__global__ void pool_sum(const float* __restrict__ h, const int* __restrict__ batch,
                         float* __restrict__ sums, float* __restrict__ counts, int nN, int F) {
    int i = blockIdx.x * blockDim.x + threadIdx.x;
    if (i >= nN * F) return;
    int n = i / F, c = i % F;
    int g = batch[n];
    atomicAdd(&sums[(size_t)g * F + c], h[i]);
    if (c == 0) atomicAdd(&counts[g], 1.0f);
}

__global__ void heads_kernel(const float* __restrict__ sums, const float* __restrict__ counts,
                             const float* __restrict__ fyw, const float* __restrict__ fyb,
                             const float* __restrict__ fpw, const float* __restrict__ fpb,
                             const float* __restrict__ fow, const float* __restrict__ fob,
                             const float* __restrict__ wm, const float* __restrict__ wd,
                             float* __restrict__ out) {
    int g = blockIdx.x * blockDim.x + threadIdx.x;
    if (g >= NGRAPH) return;
    float pooled[HIDC];
    float cnt = fmaxf(counts[g], 1.0f);
    for (int c = 0; c < HIDC; ++c) pooled[c] = sums[(size_t)g * HIDC + c] / cnt;
    float ymrs = fyb[0], phq9 = fpb[0];
    for (int c = 0; c < HIDC; ++c) { ymrs += pooled[c] * fyw[c]; phq9 += pooled[c] * fpw[c]; }
    float z[OUTC];
    for (int j = 0; j < OUTC; ++j) z[j] = fob[j];
    for (int c = 0; c < HIDC; ++c)
        for (int j = 0; j < OUTC; ++j) z[j] += pooled[c] * fow[c * OUTC + j];
    float mx = z[0];
    for (int j = 1; j < OUTC; ++j) mx = fmaxf(mx, z[j]);
    float sum = 0.f, ez[OUTC];
    for (int j = 0; j < OUTC; ++j) { ez[j] = __expf(z[j] - mx); sum += ez[j]; }
    for (int j = 0; j < OUTC; ++j) out[g * OUTC + j] = ez[j] / sum;   // logits [64,4]
    out[NGRAPH * OUTC + g]              = ymrs;                        // ymrs [64]
    out[NGRAPH * OUTC + NGRAPH + g]     = phq9;                        // phq9 [64]
    out[NGRAPH * OUTC + 2 * NGRAPH + g] = wm[0] * ymrs - wd[0] * phq9; // csm  [64]
}

// ---------------- launcher ----------------
static inline int cdiv(long long a, long long b) { return (int)((a + b - 1) / b); }

extern "C" void kernel_launch(void* const* d_in, const int* in_sizes, int n_in,
                              void* d_out, int out_size, void* d_ws, size_t ws_size,
                              hipStream_t stream) {
    const float* x        = (const float*)d_in[0];
    const int*   ei       = (const int*)  d_in[1];
    const int*   batch    = (const int*)  d_in[2];
    const float* W1       = (const float*)d_in[3];
    const float* att_src1 = (const float*)d_in[4];
    const float* att_dst1 = (const float*)d_in[5];
    const float* b1       = (const float*)d_in[6];
    const float* W2       = (const float*)d_in[7];
    const float* att_src2 = (const float*)d_in[8];
    const float* att_dst2 = (const float*)d_in[9];
    const float* b2       = (const float*)d_in[10];
    const float* fyw      = (const float*)d_in[11];
    const float* fyb      = (const float*)d_in[12];
    const float* fpw      = (const float*)d_in[13];
    const float* fpb      = (const float*)d_in[14];
    const float* fow      = (const float*)d_in[15];
    const float* fob      = (const float*)d_in[16];
    const float* wm       = (const float*)d_in[17];
    const float* wd       = (const float*)d_in[18];

    const int E = in_sizes[1] / 2;
    const int Etot = E + NNODES;

    // workspace arena (256B aligned slices)
    char* ws = (char*)d_ws;
    size_t off = 0;
    auto arena = [&](size_t bytes) -> void* {
        off = (off + 255) & ~(size_t)255;
        void* p = ws + off;
        off += bytes;
        return p;
    };
    __bf16* xb   = (__bf16*)arena((size_t)NNODES * INC * 2);
    __bf16* w1t  = (__bf16*)arena((size_t)INC * H1OUT * 2);     // [256,128]
    __bf16* w2t  = (__bf16*)arena((size_t)H1OUT * HIDC * 2);    // [64,256]
    float*  h1   = (float*) arena((size_t)NNODES * H1OUT * 4);
    float*  as1  = (float*) arena((size_t)NNODES * NHEADS * 4);
    float*  ad1  = (float*) arena((size_t)NNODES * NHEADS * 4);
    int*    m1   = (int*)   arena((size_t)NNODES * NHEADS * 4);
    float*  dn1  = (float*) arena((size_t)NNODES * NHEADS * 4);
    float*  agg1 = (float*) arena((size_t)NNODES * H1OUT * 4);
    __bf16* h1b  = (__bf16*)arena((size_t)NNODES * H1OUT * 2);
    float*  h2   = (float*) arena((size_t)NNODES * HIDC * 4);
    float*  as2  = (float*) arena((size_t)NNODES * 4);
    float*  ad2  = (float*) arena((size_t)NNODES * 4);
    int*    m2   = (int*)   arena((size_t)NNODES * 4);
    float*  dn2  = (float*) arena((size_t)NNODES * 4);
    float*  agg2 = (float*) arena((size_t)NNODES * HIDC * 4);
    float*  h2a  = (float*) arena((size_t)NNODES * HIDC * 4);
    float*  sums = (float*) arena((size_t)NGRAPH * HIDC * 4);
    float*  cnts = (float*) arena((size_t)NGRAPH * 4);
    (void)ws_size;

    const int T = 256;

    // prep: casts + weight transposes
    cast_f32_bf16<<<cdiv((long long)NNODES * INC, T), T, 0, stream>>>(x, xb, NNODES * INC);
    transpose_cast<<<cdiv(INC * H1OUT, T), T, 0, stream>>>(W1, w1t, INC, H1OUT);
    transpose_cast<<<cdiv(H1OUT * HIDC, T), T, 0, stream>>>(W2, w2t, H1OUT, HIDC);

    // ---- layer 1 ----
    {
        int tiles = (NNODES / 16) * (H1OUT / 16);
        gemm_bf16_wmma<<<cdiv(tiles, 8), T, 0, stream>>>(xb, w1t, h1, NNODES, INC, H1OUT);
    }
    att_coeff<<<cdiv(NNODES * NHEADS, T), T, 0, stream>>>(h1, att_src1, att_dst1, as1, ad1,
                                                          NNODES, NHEADS, HIDC);
    fill_i32<<<cdiv(NNODES * NHEADS, T), T, 0, stream>>>(m1, ENC_NEG_INF, NNODES * NHEADS);
    fill_f32<<<cdiv(NNODES * NHEADS, T), T, 0, stream>>>(dn1, 0.f, NNODES * NHEADS);
    fill_f32<<<cdiv((long long)NNODES * H1OUT, T), T, 0, stream>>>(agg1, 0.f, NNODES * H1OUT);
    edge_max<<<cdiv((long long)Etot * NHEADS, T), T, 0, stream>>>(ei, as1, ad1, m1, E, NNODES, NHEADS);
    edge_sum<<<cdiv((long long)Etot * NHEADS, T), T, 0, stream>>>(ei, as1, ad1, m1, dn1, E, NNODES, NHEADS);
    edge_agg<<<cdiv((long long)Etot * NHEADS, T), T, 0, stream>>>(ei, as1, ad1, m1, dn1, h1, agg1,
                                                                  E, NNODES, NHEADS, HIDC);
    bias_elu<<<cdiv((long long)NNODES * H1OUT, T), T, 0, stream>>>(agg1, b1, (float*)nullptr, h1b,
                                                                   NNODES, H1OUT);

    // ---- layer 2 ----
    {
        int tiles = (NNODES / 16) * (HIDC / 16);
        gemm_bf16_wmma<<<cdiv(tiles, 8), T, 0, stream>>>(h1b, w2t, h2, NNODES, H1OUT, HIDC);
    }
    att_coeff<<<cdiv(NNODES, T), T, 0, stream>>>(h2, att_src2, att_dst2, as2, ad2, NNODES, 1, HIDC);
    fill_i32<<<cdiv(NNODES, T), T, 0, stream>>>(m2, ENC_NEG_INF, NNODES);
    fill_f32<<<cdiv(NNODES, T), T, 0, stream>>>(dn2, 0.f, NNODES);
    fill_f32<<<cdiv((long long)NNODES * HIDC, T), T, 0, stream>>>(agg2, 0.f, NNODES * HIDC);
    edge_max<<<cdiv(Etot, T), T, 0, stream>>>(ei, as2, ad2, m2, E, NNODES, 1);
    edge_sum<<<cdiv(Etot, T), T, 0, stream>>>(ei, as2, ad2, m2, dn2, E, NNODES, 1);
    edge_agg<<<cdiv(Etot, T), T, 0, stream>>>(ei, as2, ad2, m2, dn2, h2, agg2, E, NNODES, 1, HIDC);
    bias_elu<<<cdiv((long long)NNODES * HIDC, T), T, 0, stream>>>(agg2, b2, h2a, (__bf16*)nullptr,
                                                                  NNODES, HIDC);

    // ---- pooling + heads ----
    fill_f32<<<cdiv(NGRAPH * HIDC, T), T, 0, stream>>>(sums, 0.f, NGRAPH * HIDC);
    fill_f32<<<1, 64, 0, stream>>>(cnts, 0.f, NGRAPH);
    pool_sum<<<cdiv((long long)NNODES * HIDC, T), T, 0, stream>>>(h2a, batch, sums, cnts, NNODES, HIDC);
    heads_kernel<<<1, 64, 0, stream>>>(sums, cnts, fyw, fyb, fpw, fpb, fow, fob, wm, wd, (float*)d_out);
}
